// LowRankNextCharacterLSTM_60679297958107
// MI455X (gfx1250) — matrix operand order repaired and checked by reading, not compiled, not run on hardware
//
#include <hip/hip_runtime.h>
#include <cstdint>

// ---------------- problem constants ----------------
#define T_SEQ 256
#define BATCH 64
#define HID   1024
#define RG    1024   // low-rank gate dim
#define G4    4096   // 4*HID
#define VOC   256
#define RD    64     // low-rank decoder dim
#define NLAYER 2

typedef __bf16 bf16;
typedef __bf16 v16bf __attribute__((ext_vector_type(16)));
typedef float  v8f   __attribute__((ext_vector_type(8)));

// ---------------- WMMA fragment loaders (ISA 7.12.2 layouts, wave32) -------
// A (16x32 bf16): lanes 0-15 -> M=lane, K chunks [0,8)+[16,24);
//                 lanes 16-31 -> M=lane-16, K chunks [8,16)+[24,32).
static __device__ inline v16bf load_frag_a(const bf16* __restrict__ tile,
                                           int ldk, int lane) {
  const int m   = lane & 15;
  const int off = (lane >> 4) << 3;       // 0 or 8
  const bf16* p = tile + m * ldk + off;
  v16bf r;
#pragma unroll
  for (int i = 0; i < 8; ++i) r[i]     = p[i];
#pragma unroll
  for (int i = 0; i < 8; ++i) r[8 + i] = p[16 + i];
  return r;
}

// B (32x16 bf16, B = W^T with W [N,K] row-major):
// lanes 0-15 -> column n=lane, K chunks [0,8)+[16,24); lanes 16-31 -> n=lane-16, K chunks [8,16)+[24,32).
static __device__ inline v16bf load_frag_b(const bf16* __restrict__ w,
                                           long ldk, int lane) {
  const int n   = lane & 15;
  const int off = (lane >> 4) << 3;       // 0 or 8
  const bf16* p = w + (long)n * ldk + off;
  v16bf r;
#pragma unroll
  for (int i = 0; i < 8; ++i) r[i]     = p[i];
#pragma unroll
  for (int i = 0; i < 8; ++i) r[8 + i] = p[16 + i];
  return r;
}

static __device__ inline v8f wmma_bf16(v16bf a, v16bf b, v8f c) {
  return __builtin_amdgcn_wmma_f32_16x16x32_bf16(
      /*neg_a=*/false, a, /*neg_b=*/false, b,
      /*c_mod=*/(short)0, c, /*reuse_a=*/false, /*reuse_b=*/false);
}

// ---------------- generic WMMA GEMM: C[M,N] = A[M,K] @ W[N,K]^T + bias[N] --
// Block = 8 waves sharing one 16-row A tile (LDS). Each wave owns a 16x64
// output strip: 4 independent fp32 accumulators -> 4 independent WMMAs per
// K-chunk pipelined on the XDL pipe, A fragment loaded once per chunk.
// OMODE 1: bf16 row-major output.  OMODE 2: f32 output transposed to [B,V,T].
template <int OMODE>
__global__ __launch_bounds__(256) void gemm_k(
    const bf16* __restrict__ A, const bf16* __restrict__ W,
    const float* __restrict__ bias, void* __restrict__ Cout,
    int N, int K) {
  __shared__ bf16 sA[16 * 32];
  const int lane    = threadIdx.x & 31;
  const int wave    = threadIdx.x >> 5;
  const int row0    = blockIdx.y << 4;
  const int nstrip0 = ((blockIdx.x << 3) + wave) << 6;   // 64-wide N strip
  const bool active = nstrip0 < N;                        // N % 64 == 0
  const bf16* wp = W + (long)nstrip0 * K;

  const int lr = threadIdx.x >> 4;          // 0..15 row of A tile
  const int lk = (threadIdx.x & 15) << 1;   // even K offset (one dword)
  const bf16* arow = A + (long)(row0 + lr) * K + lk;

  v8f acc[4] = {};
  for (int k0 = 0; k0 < K; k0 += 32) {
    // cooperative 1KB A-tile stage: 256 threads x 1 dword
    *(uint32_t*)(&sA[lr * 32 + lk]) = *(const uint32_t*)(arow + k0);
    __syncthreads();
    if (active) {
      v16bf af = load_frag_a(sA, 32, lane);          // ds_load, reused 4x
#pragma unroll
      for (int i = 0; i < 4; ++i) {
        const bf16* wpi = wp + (long)(i << 4) * K + k0;
        __builtin_prefetch(wpi + 32, 0, 3);          // global_prefetch next K
        v16bf bf = load_frag_b(wpi, K, lane);        // global_load from L2
        acc[i] = wmma_bf16(af, bf, acc[i]);          // 4 independent WMMAs
      }
    }
    __syncthreads();
  }
  if (!active) return;

  // C layout: VGPR r, lane l -> row = row0 + 8*(l>=16) + r, col = n0 + (l&15)
  const int rb = row0 + ((lane >> 4) << 3);
#pragma unroll
  for (int i = 0; i < 4; ++i) {
    const int cn = nstrip0 + (i << 4) + (lane & 15);
    const float bv = bias[cn];
    if (OMODE == 1) {
      bf16* C = (bf16*)Cout;
#pragma unroll
      for (int r = 0; r < 8; ++r)
        C[(long)(rb + r) * N + cn] = (bf16)(acc[i][r] + bv);
    } else {
      // rows are (t*BATCH + b); logits out[b, v, t]
      float* C = (float*)Cout;
#pragma unroll
      for (int r = 0; r < 8; ++r) {
        const int m = rb + r;
        C[((long)(m & (BATCH - 1)) * N + cn) * T_SEQ + (m >> 6)] =
            acc[i][r] + bv;
      }
    }
  }
}

// ------- fused dual-K GEMM (recurrent step): G = A1@W1^T + A2@W2^T + b1 + b2
// Same 16x64-strip-per-wave scheme; 8 WMMAs per K-chunk, 2 chained per accumulator.
__global__ __launch_bounds__(256) void gemm_dual_k(
    const bf16* __restrict__ A1, const bf16* __restrict__ W1,
    const float* __restrict__ b1,
    const bf16* __restrict__ A2, const bf16* __restrict__ W2,
    const float* __restrict__ b2,
    float* __restrict__ G, int N, int K) {
  __shared__ bf16 sA1[16 * 32];
  __shared__ bf16 sA2[16 * 32];
  const int lane    = threadIdx.x & 31;
  const int wave    = threadIdx.x >> 5;
  const int row0    = blockIdx.y << 4;
  const int nstrip0 = ((blockIdx.x << 3) + wave) << 6;
  const bf16* w1 = W1 + (long)nstrip0 * K;
  const bf16* w2 = W2 + (long)nstrip0 * K;

  const int lr = threadIdx.x >> 4;
  const int lk = (threadIdx.x & 15) << 1;
  const bf16* a1row = A1 + (long)(row0 + lr) * K + lk;
  const bf16* a2row = A2 + (long)(row0 + lr) * K + lk;

  v8f acc[4] = {};
  for (int k0 = 0; k0 < K; k0 += 32) {
    *(uint32_t*)(&sA1[lr * 32 + lk]) = *(const uint32_t*)(a1row + k0);
    *(uint32_t*)(&sA2[lr * 32 + lk]) = *(const uint32_t*)(a2row + k0);
    __syncthreads();
    v16bf a1f = load_frag_a(sA1, 32, lane);
    v16bf a2f = load_frag_a(sA2, 32, lane);
#pragma unroll
    for (int i = 0; i < 4; ++i) {
      const bf16* w1i = w1 + (long)(i << 4) * K + k0;
      const bf16* w2i = w2 + (long)(i << 4) * K + k0;
      __builtin_prefetch(w1i + 32, 0, 3);
      __builtin_prefetch(w2i + 32, 0, 3);
      v16bf b1f = load_frag_b(w1i, K, lane);
      acc[i] = wmma_bf16(a1f, b1f, acc[i]);
      v16bf b2f = load_frag_b(w2i, K, lane);
      acc[i] = wmma_bf16(a2f, b2f, acc[i]);
    }
    __syncthreads();
  }
  const int rb = row0 + ((lane >> 4) << 3);
#pragma unroll
  for (int i = 0; i < 4; ++i) {
    const int cn = nstrip0 + (i << 4) + (lane & 15);
    const float bv = b1[cn] + b2[cn];
#pragma unroll
    for (int r = 0; r < 8; ++r)
      G[(long)(rb + r) * N + cn] = acc[i][r] + bv;
  }
}

// ---------------- LSTM cell elementwise update -----------------------------
__global__ __launch_bounds__(256) void lstm_cell_k(
    const float* __restrict__ G, float* __restrict__ c,
    bf16* __restrict__ hbf, bf16* __restrict__ Yt) {
  const int i = blockIdx.x * 256 + threadIdx.x;   // 0 .. BATCH*HID-1
  const int b = i >> 10;
  const int j = i & (HID - 1);
  const float* g = G + (long)b * G4;
  const float xi = g[j];
  const float xf = g[HID + j];
  const float xg = g[2 * HID + j];
  const float xo = g[3 * HID + j];
  const float si = 1.f / (1.f + __expf(-xi));
  const float sf = 1.f / (1.f + __expf(-xf));
  const float tg = tanhf(xg);
  const float so = 1.f / (1.f + __expf(-xo));
  const float cn = sf * c[i] + si * tg;
  const float h  = so * tanhf(cn);
  c[i] = cn;
  const bf16 hb = (bf16)h;
  hbf[i] = hb;      // recurrent input for next step
  Yt[i]  = hb;      // sequence output row t (rows are t*BATCH+b)
}

// ---------------- small helpers --------------------------------------------
__global__ __launch_bounds__(256) void f32_to_bf16_k(
    const float* __restrict__ s, bf16* __restrict__ d, int n) {
  const int i = blockIdx.x * 256 + threadIdx.x;
  if (i < n) d[i] = (bf16)s[i];
}

__global__ __launch_bounds__(256) void init_state_k(
    bf16* __restrict__ hbf, float* __restrict__ c) {
  const int i = blockIdx.x * 256 + threadIdx.x;
  hbf[i] = (bf16)0.f;
  c[i] = 0.f;
}

// X[t*BATCH+b, e] = bf16(embed[tokens[b, t], e])
__global__ __launch_bounds__(256) void embed_k(
    const int* __restrict__ tokens, const float* __restrict__ emb,
    bf16* __restrict__ X) {
  const long i = (long)blockIdx.x * 256 + threadIdx.x;
  const int e  = (int)(i & (HID - 1));
  const int tb = (int)(i >> 10);
  const int b  = tb & (BATCH - 1);
  const int t  = tb >> 6;
  const int tok = tokens[b * T_SEQ + t];
  X[i] = (bf16)emb[(long)tok * HID + e];
}

// ---------------- host driver ----------------------------------------------
extern "C" void kernel_launch(void* const* d_in, const int* in_sizes, int n_in,
                              void* d_out, int out_size, void* d_ws,
                              size_t ws_size, hipStream_t stream) {
  (void)in_sizes; (void)n_in; (void)out_size; (void)ws_size;
  const int*   tokens = (const int*)d_in[0];
  const float* embedW = (const float*)d_in[1];
  const float* VT_ih  = (const float*)d_in[2];
  const float* bVT_ih = (const float*)d_in[3];
  const float* U_ih   = (const float*)d_in[4];
  const float* bU_ih  = (const float*)d_in[5];
  const float* VT_hh  = (const float*)d_in[6];
  const float* bVT_hh = (const float*)d_in[7];
  const float* U_hh   = (const float*)d_in[8];
  const float* bU_hh  = (const float*)d_in[9];
  const float* dec_VT = (const float*)d_in[10];
  const float* dec_bVT= (const float*)d_in[11];
  const float* dec_U  = (const float*)d_in[12];
  const float* dec_bU = (const float*)d_in[13];

  // ---- workspace carve (all offsets 256B aligned) ----
  char* p = (char*)d_ws;
  auto take = [&](size_t bytes) {
    char* q = p; p += (bytes + 255) & ~(size_t)255; return q;
  };
  bf16* Xbf   = (bf16*)take((size_t)T_SEQ * BATCH * HID * 2);  // layer input
  bf16* Ybf   = (bf16*)take((size_t)T_SEQ * BATCH * HID * 2);  // layer output
  bf16* Px    = (bf16*)take((size_t)T_SEQ * BATCH * RG * 2);   // hoisted x-path
  bf16* wVTih = (bf16*)take((size_t)NLAYER * RG * HID * 2);
  bf16* wUih  = (bf16*)take((size_t)NLAYER * G4 * RG * 2);
  bf16* wVThh = (bf16*)take((size_t)NLAYER * RG * HID * 2);
  bf16* wUhh  = (bf16*)take((size_t)NLAYER * G4 * RG * 2);
  bf16* wdVT  = (bf16*)take((size_t)RD * HID * 2);
  bf16* wdU   = (bf16*)take((size_t)VOC * RD * 2);
  bf16* hbf   = (bf16*)take((size_t)BATCH * HID * 2);
  float* cst  = (float*)take((size_t)BATCH * HID * 4);
  bf16* Ph    = (bf16*)take((size_t)BATCH * RG * 2);
  float* G    = (float*)take((size_t)BATCH * G4 * 4);
  bf16* Pd    = (bf16*)take((size_t)T_SEQ * BATCH * RD * 2);

  // ---- weights -> bf16 (L2-resident working set ~40 MB) ----
  auto conv = [&](const float* s, bf16* d, size_t n) {
    f32_to_bf16_k<<<dim3((unsigned)((n + 255) / 256)), dim3(256), 0, stream>>>(
        s, d, (int)n);
  };
  conv(VT_ih, wVTih, (size_t)NLAYER * RG * HID);
  conv(U_ih,  wUih,  (size_t)NLAYER * G4 * RG);
  conv(VT_hh, wVThh, (size_t)NLAYER * RG * HID);
  conv(U_hh,  wUhh,  (size_t)NLAYER * G4 * RG);
  conv(dec_VT, wdVT, (size_t)RD * HID);
  conv(dec_U,  wdU,  (size_t)VOC * RD);

  // ---- embedding gather, time-major ----
  embed_k<<<dim3((unsigned)((size_t)T_SEQ * BATCH * HID / 256)), dim3(256), 0,
            stream>>>(tokens, embedW, Xbf);

  // blocks cover 512 output columns (8 waves x 64-wide strips)
  auto nblk = [](int N) { return (unsigned)((N + 511) / 512); };

  // ---- layers ----
  bf16* Xcur = Xbf;
  bf16* Ycur = Ybf;
  for (int l = 0; l < NLAYER; ++l) {
    const bf16* lVTih = wVTih + (size_t)l * RG * HID;
    const bf16* lUih  = wUih  + (size_t)l * G4 * RG;
    const bf16* lVThh = wVThh + (size_t)l * RG * HID;
    const bf16* lUhh  = wUhh  + (size_t)l * G4 * RG;
    const float* lbVTih = bVT_ih + (size_t)l * RG;
    const float* lbUih  = bU_ih  + (size_t)l * G4;
    const float* lbVThh = bVT_hh + (size_t)l * RG;
    const float* lbUhh  = bU_hh  + (size_t)l * G4;

    // hoisted x-path: Px = X @ VT_ih^T + bVT_ih   [16384 x 1024, K=1024]
    gemm_k<1><<<dim3(nblk(RG), T_SEQ * BATCH / 16), dim3(256), 0, stream>>>(
        Xcur, lVTih, lbVTih, Px, RG, HID);

    init_state_k<<<dim3(BATCH * HID / 256), dim3(256), 0, stream>>>(hbf, cst);

    for (int t = 0; t < T_SEQ; ++t) {
      // Ph = h @ VT_hh^T + bVT_hh      [64 x 1024, K=1024]
      gemm_k<1><<<dim3(nblk(RG), BATCH / 16), dim3(256), 0, stream>>>(
          hbf, lVThh, lbVThh, Ph, RG, HID);
      // gates = Ph @ U_hh^T + Px[t] @ U_ih^T + bU_hh + bU_ih  [64 x 4096]
      gemm_dual_k<<<dim3(nblk(G4), BATCH / 16), dim3(256), 0, stream>>>(
          Ph, lUhh, lbUhh, Px + (size_t)t * BATCH * RG, lUih, lbUih, G, G4, RG);
      // elementwise cell; writes h (bf16) for recurrence and sequence output
      lstm_cell_k<<<dim3(BATCH * HID / 256), dim3(256), 0, stream>>>(
          G, cst, hbf, Ycur + (size_t)t * BATCH * HID);
    }
    bf16* tmp = Xcur; Xcur = Ycur; Ycur = tmp;  // next layer input = outputs
  }

  // ---- low-rank decoder ----
  // Pd = Y @ dec_VT^T + dec_bVT   [16384 x 64, K=1024]
  gemm_k<1><<<dim3(nblk(RD), T_SEQ * BATCH / 16), dim3(256), 0, stream>>>(
      Xcur, wdVT, dec_bVT, Pd, RD, HID);
  // logits = Pd @ dec_U^T + dec_bU, written transposed to out[B, V, T]
  gemm_k<2><<<dim3(nblk(VOC), T_SEQ * BATCH / 16), dim3(256), 0, stream>>>(
      Pd, wdU, dec_bU, (float*)d_out, VOC, RD);
}
